// SlidingWindowAttention_49125835932010
// MI455X (gfx1250) — compile-verified, matrix operand order checked by reading, not run on hardware
//
#include <hip/hip_runtime.h>
#include <hip/hip_bf16.h>
#include <math.h>

// ---------------------------------------------------------------------------
// Sliding-window attention for MI455X (gfx1250), wave32 + WMMA f16.
//   B=4 S=4096 E=1024 H=16 D=64 W=256 Nw=16
// ---------------------------------------------------------------------------

typedef _Float16 h16v __attribute__((ext_vector_type(16)));
typedef _Float16 h8v  __attribute__((ext_vector_type(8)));
typedef float    f8v  __attribute__((ext_vector_type(8)));

#define WMMA_F16(A, B, C) \
    __builtin_amdgcn_wmma_f32_16x16x32_f16(false, (A), false, (B), (short)0, (C), false, false)

// Load one 16x16x32 A/B fragment for lane `lane`:
//   lane holds row (lane&15), K = kbase..kbase+7 and kbase+16..kbase+23,
//   where kbase = (lane>>4)*8. Caller passes p = &row[kbase].
// Two 16-byte loads -> 8 VGPRs. Works for global or LDS pointers.
__device__ inline h16v load_frag(const _Float16* __restrict__ p) {
    h8v lo = *(const h8v*)(p);
    h8v hi = *(const h8v*)(p + 16);
    return __builtin_shufflevector(lo, hi, 0, 1, 2, 3, 4, 5, 6, 7,
                                   8, 9, 10, 11, 12, 13, 14, 15);
}

__device__ inline float half16_max(float v) {
    // max across each 16-lane half (rows stay within halves in C layout)
    v = fmaxf(v, __shfl_xor(v, 1, 32));
    v = fmaxf(v, __shfl_xor(v, 2, 32));
    v = fmaxf(v, __shfl_xor(v, 4, 32));
    v = fmaxf(v, __shfl_xor(v, 8, 32));
    return v;
}

// ---------------------------------------------------------------------------
// Kernel 1: f32 -> f16 conversion
// ---------------------------------------------------------------------------
__global__ __launch_bounds__(256) void cvt_f32_to_f16(const float* __restrict__ src,
                                                      _Float16* __restrict__ dst, int n) {
    int i = blockIdx.x * 256 + threadIdx.x;
    if (i < n) dst[i] = (_Float16)src[i];
}

// ---------------------------------------------------------------------------
// Kernel 2: QKV projection  qkv = x @ W_qkv^T + b_qkv,  scatter into
//   q,k : [B,H,S,D] row-major      vt : [B,H,D,S] (transposed V)
// M=16384 (B*S), N=3072 (3E), K=1024.  WG = 256 thr = 8 waves (2x4),
// each wave computes a 64x64 tile = 4x4 WMMA accumulators.
// ---------------------------------------------------------------------------
__global__ __launch_bounds__(256) void qkv_gemm(const _Float16* __restrict__ xh,
                                                const _Float16* __restrict__ wh,
                                                const float* __restrict__ bias,
                                                _Float16* __restrict__ qo,
                                                _Float16* __restrict__ ko,
                                                _Float16* __restrict__ vt) {
    const int lane = threadIdx.x & 31;
    const int wid  = threadIdx.x >> 5;
    const int lr   = lane & 15;
    const int lh   = lane >> 4;
    const int m0   = blockIdx.x * 128 + (wid >> 2) * 64;
    const int n0   = blockIdx.y * 256 + (wid & 3) * 64;

    f8v acc[4][4] = {};

    for (int kt = 0; kt < 32; ++kt) {
        const int kb = kt * 32 + lh * 8;
        h16v a[4], b[4];
#pragma unroll
        for (int i = 0; i < 4; ++i)
            a[i] = load_frag(xh + (size_t)(m0 + i * 16 + lr) * 1024 + kb);
#pragma unroll
        for (int j = 0; j < 4; ++j)
            b[j] = load_frag(wh + (size_t)(n0 + j * 16 + lr) * 1024 + kb);
#pragma unroll
        for (int i = 0; i < 4; ++i)
#pragma unroll
            for (int j = 0; j < 4; ++j)
                acc[i][j] = WMMA_F16(a[i], b[j], acc[i][j]);
    }

    // Epilogue: bias + scatter.  C layout: VGPR r -> (M = r + 8*lh, N = lr).
#pragma unroll
    for (int j = 0; j < 4; ++j) {
        const int col = n0 + j * 16 + lr;     // 0..3071
        const float bv = bias[col];
        const int sec = col >> 10;            // 0=q 1=k 2=v
        const int c   = col & 1023;
        const int h   = c >> 6;
        const int d   = c & 63;
#pragma unroll
        for (int i = 0; i < 4; ++i) {
            if (sec == 2) {
                // V section: for a fixed lane, r = 0..7 are 8 consecutive s
                // positions of one d-row of vt -> pack into one 16B store.
                const int m0i = m0 + i * 16 + lh * 8;   // first s of this run
                const int bb  = m0i >> 12;
                const int s0  = m0i & 4095;
                h8v pk;
#pragma unroll
                for (int r = 0; r < 8; ++r)
                    pk[r] = (_Float16)(acc[i][j][r] + bv);
                *(h8v*)(vt + ((size_t)(bb * 16 + h) * 64 + d) * 4096 + s0) = pk;
            } else {
#pragma unroll
                for (int r = 0; r < 8; ++r) {
                    const int m  = m0 + i * 16 + r + lh * 8;  // row = b*S + s
                    const int bb = m >> 12;
                    const int s  = m & 4095;
                    const _Float16 hv = (_Float16)(acc[i][j][r] + bv);
                    // lanes lr=0..15 write consecutive d -> coalesced 32B runs
                    if (sec == 0)
                        qo[(((size_t)(bb * 16 + h) * 4096 + s) << 6) + d] = hv;
                    else
                        ko[(((size_t)(bb * 16 + h) * 4096 + s) << 6) + d] = hv;
                }
            }
        }
    }
}

// ---------------------------------------------------------------------------
// Kernel 3: windowed attention. One WG (256 thr, 8 waves) per (b,h,window).
//   Phase 1: S = (q @ k^T) * 0.125 -> f16 scores in LDS, row max in regs
//   Phase 2: thread-per-row: vectorized exp (unnormalized) + 1/rowsum
//   Phase 3: O = (P @ V) * inv_rowsum via WMMA
// LDS: 256 rows * 264 f16 (stride-padded) + 256 f32 row stats.
// ---------------------------------------------------------------------------
#define SROW 264  // f16 elements per score row (256 + pad); 528 B, 16B aligned

__global__ __launch_bounds__(256) void attn_win(const _Float16* __restrict__ q,
                                                const _Float16* __restrict__ k,
                                                const _Float16* __restrict__ vt,
                                                _Float16* __restrict__ attn) {
    extern __shared__ char smem_raw[];
    _Float16* sc = (_Float16*)smem_raw;                     // [256][SROW]
    float* rowred = (float*)(smem_raw + 256 * SROW * 2);    // [256] max, then 1/sum

    const int wg = blockIdx.x;       // b*H*Nw
    const int w  = wg & 15;          // window index
    const int bh = wg >> 4;          // b*H + h
    const int lane = threadIdx.x & 31;
    const int wid  = threadIdx.x >> 5;
    const int lr = lane & 15, lh = lane >> 4;
    const int lm0 = wid * 32;        // this wave's 32 query rows

    const _Float16* qbase = q + ((size_t)bh * 4096 + w * 256) * 64;
    const _Float16* kbase = k + ((size_t)bh * 4096 + w * 256) * 64;
    const _Float16* vbase = vt + (size_t)bh * 64 * 4096 + w * 256;

    // Resident q fragments: 2 M-subtiles x 2 K-steps (K=64)
    h16v qa[2][2];
#pragma unroll
    for (int m = 0; m < 2; ++m)
#pragma unroll
        for (int ks = 0; ks < 2; ++ks)
            qa[m][ks] = load_frag(qbase + (size_t)(lm0 + m * 16 + lr) * 64 + ks * 32 + lh * 8);

    float rmax[2][8];
#pragma unroll
    for (int m = 0; m < 2; ++m)
#pragma unroll
        for (int r = 0; r < 8; ++r) rmax[m][r] = -3.0e38f;

    // ---- Phase 1: scores ----
    for (int nt = 0; nt < 16; ++nt) {
        f8v accs[2] = {};
#pragma unroll
        for (int ks = 0; ks < 2; ++ks) {
            h16v kf = load_frag(kbase + (size_t)(nt * 16 + lr) * 64 + ks * 32 + lh * 8);
            accs[0] = WMMA_F16(qa[0][ks], kf, accs[0]);
            accs[1] = WMMA_F16(qa[1][ks], kf, accs[1]);
        }
#pragma unroll
        for (int m = 0; m < 2; ++m) {
#pragma unroll
            for (int r = 0; r < 8; ++r) {
                const float v = accs[m][r] * 0.125f;  // 1/sqrt(64)
                rmax[m][r] = fmaxf(rmax[m][r], half16_max(v));
                const int row = lm0 + m * 16 + r + lh * 8;
                sc[row * SROW + nt * 16 + lr] = (_Float16)v;
            }
        }
    }
#pragma unroll
    for (int m = 0; m < 2; ++m)
#pragma unroll
        for (int r = 0; r < 8; ++r)
            if (lr == 0) rowred[lm0 + m * 16 + r + lh * 8] = rmax[m][r];
    __syncthreads();

    // ---- Phase 2: one thread per row; single vectorized exp pass ----
    {
        const int row = threadIdx.x;
        _Float16* sr = sc + row * SROW;
        const float rm = rowred[row];
        float sum = 0.0f;
        for (int c0 = 0; c0 < 256; c0 += 8) {
            h8v vv = *(h8v*)(sr + c0);
            h8v ev;
#pragma unroll
            for (int u = 0; u < 8; ++u) {
                const float e = __expf((float)vv[u] - rm);
                sum += e;
                ev[u] = (_Float16)e;
            }
            *(h8v*)(sr + c0) = ev;   // unnormalized probabilities
        }
        rowred[row] = 1.0f / sum;    // overwrite max with reciprocal row-sum
    }
    __syncthreads();

    // ---- Phase 3: O = (P @ V) * inv_rowsum ----
    f8v oacc[2][4] = {};
    for (int kstep = 0; kstep < 8; ++kstep) {
        const int kk = kstep * 32;
        h16v pa[2];
#pragma unroll
        for (int m = 0; m < 2; ++m)
            pa[m] = load_frag(sc + (size_t)(lm0 + m * 16 + lr) * SROW + kk + lh * 8);
#pragma unroll
        for (int j = 0; j < 4; ++j) {
            h16v vf = load_frag(vbase + (size_t)(j * 16 + lr) * 4096 + kk + lh * 8);
            oacc[0][j] = WMMA_F16(pa[0], vf, oacc[0][j]);
            oacc[1][j] = WMMA_F16(pa[1], vf, oacc[1][j]);
        }
    }

    const int b_ = bh >> 4, h_ = bh & 15;
#pragma unroll
    for (int m = 0; m < 2; ++m) {
#pragma unroll
        for (int r = 0; r < 8; ++r) {
            const int rowl = lm0 + m * 16 + r + lh * 8;   // local query row
            const float inv = rowred[rowl];               // per-lane row factor
            const int sg = w * 256 + rowl;
#pragma unroll
            for (int j = 0; j < 4; ++j) {
                const int d = j * 16 + lr;
                // lanes lr=0..15 write consecutive d -> coalesced 32B runs
                attn[((size_t)(b_ * 4096 + sg)) * 1024 + h_ * 64 + d] =
                    (_Float16)(oacc[m][j][r] * inv);
            }
        }
    }
}

// ---------------------------------------------------------------------------
// Kernel 4: output projection  out = attn @ W_out^T + b_out  (f32 output)
// M=16384, N=1024, K=1024, same tiling as kernel 2.
// ---------------------------------------------------------------------------
__global__ __launch_bounds__(256) void out_gemm(const _Float16* __restrict__ ah,
                                                const _Float16* __restrict__ wh,
                                                const float* __restrict__ bias,
                                                float* __restrict__ out) {
    const int lane = threadIdx.x & 31;
    const int wid  = threadIdx.x >> 5;
    const int lr = lane & 15, lh = lane >> 4;
    const int m0 = blockIdx.x * 128 + (wid >> 2) * 64;
    const int n0 = blockIdx.y * 256 + (wid & 3) * 64;

    f8v acc[4][4] = {};
    for (int kt = 0; kt < 32; ++kt) {
        const int kb = kt * 32 + lh * 8;
        h16v a[4], b[4];
#pragma unroll
        for (int i = 0; i < 4; ++i)
            a[i] = load_frag(ah + (size_t)(m0 + i * 16 + lr) * 1024 + kb);
#pragma unroll
        for (int j = 0; j < 4; ++j)
            b[j] = load_frag(wh + (size_t)(n0 + j * 16 + lr) * 1024 + kb);
#pragma unroll
        for (int i = 0; i < 4; ++i)
#pragma unroll
            for (int j = 0; j < 4; ++j)
                acc[i][j] = WMMA_F16(a[i], b[j], acc[i][j]);
    }
#pragma unroll
    for (int j = 0; j < 4; ++j) {
        const int col = n0 + j * 16 + lr;
        const float bv = bias[col];
#pragma unroll
        for (int i = 0; i < 4; ++i)
#pragma unroll
            for (int r = 0; r < 8; ++r) {
                const int m = m0 + i * 16 + r + lh * 8;
                out[(size_t)m * 1024 + col] = acc[i][j][r] + bv;   // coalesced
            }
    }
}

// ---------------------------------------------------------------------------
// Host launch
// ---------------------------------------------------------------------------
extern "C" void kernel_launch(void* const* d_in, const int* in_sizes, int n_in,
                              void* d_out, int out_size, void* d_ws, size_t ws_size,
                              hipStream_t stream) {
    const float* x     = (const float*)d_in[0];  // [4,4096,1024]
    const float* W_qkv = (const float*)d_in[1];  // [3072,1024]
    const float* b_qkv = (const float*)d_in[2];  // [3072]
    const float* W_out = (const float*)d_in[3];  // [1024,1024]
    const float* b_out = (const float*)d_in[4];  // [1024]

    const size_t NX   = 4u * 4096u * 1024u;  // 16,777,216
    const size_t NWQ  = 3072u * 1024u;       //  3,145,728
    const size_t NWO  = 1024u * 1024u;       //  1,048,576

    _Float16* ws    = (_Float16*)d_ws;
    _Float16* xh    = ws;            ws += NX;
    _Float16* wqkvh = ws;            ws += NWQ;
    _Float16* wouth = ws;            ws += NWO;
    _Float16* qbuf  = ws;            ws += NX;   // [B,H,S,D]
    _Float16* kbuf  = ws;            ws += NX;   // [B,H,S,D]
    _Float16* vtbuf = ws;            ws += NX;   // [B,H,D,S]
    _Float16* abuf  = ws;            ws += NX;   // [B,S,E]
    (void)ws_size; (void)in_sizes; (void)n_in; (void)out_size;

    cvt_f32_to_f16<<<(int)((NX  + 255) / 256), 256, 0, stream>>>(x,     xh,    (int)NX);
    cvt_f32_to_f16<<<(int)((NWQ + 255) / 256), 256, 0, stream>>>(W_qkv, wqkvh, (int)NWQ);
    cvt_f32_to_f16<<<(int)((NWO + 255) / 256), 256, 0, stream>>>(W_out, wouth, (int)NWO);

    qkv_gemm<<<dim3(128, 12), 256, 0, stream>>>(xh, wqkvh, b_qkv, qbuf, kbuf, vtbuf);

    const size_t lds_bytes = 256u * SROW * 2u + 256u * 4u;  // 136,192 B
    attn_win<<<dim3(1024), 256, lds_bytes, stream>>>(qbuf, kbuf, vtbuf, abuf);

    out_gemm<<<dim3(128, 4), 256, 0, stream>>>(abuf, wouth, b_out, (float*)d_out);
}
